// WorkingMemory_14594298872482
// MI455X (gfx1250) — matrix operand order repaired and checked by reading, not compiled, not run on hardware
//
#include <hip/hip_runtime.h>
#include <hip/hip_bf16.h>
#include <cstdint>
#include <cstddef>

// ---------------------------------------------------------------------------
// Problem constants (match reference)
// ---------------------------------------------------------------------------
#define BSZ   1024
#define DIM   1024
#define DWM   256
#define WSLOT 512
#define NHEAD 8
#define HDIM  32
#define ATT_SCALE 0.17677669529663687f   // 1/sqrt(32)

typedef __attribute__((ext_vector_type(16))) __bf16 v16bf;
typedef __attribute__((ext_vector_type(8)))  __bf16 v8bf;
typedef __attribute__((ext_vector_type(4)))  __bf16 v4bf;
typedef __attribute__((ext_vector_type(8)))  float  v8f;
typedef __attribute__((ext_vector_type(4)))  float  v4f;

// ---------------------------------------------------------------------------
// f32 -> bf16 conversion (weights / activations for WMMA)
// ---------------------------------------------------------------------------
__global__ void cvt_f32_bf16(const float* __restrict__ in,
                             __bf16* __restrict__ out, int n) {
  int i = blockIdx.x * blockDim.x + threadIdx.x;
  if (i < n) out[i] = (__bf16)in[i];
}

// ---------------------------------------------------------------------------
// NT GEMM via WMMA: C[M,N] = A[M,K](bf16) * B[N,K](bf16)^T + bias[N]
// One wave per 16x16 C tile, K stepped by 32 (v_wmma_f32_16x16x32_bf16).
// A fragment (16x32 bf16): lanes 0-15 hold M=0..15 with K = 0..7,16..23;
//                          lanes 16-31 hold same rows with K = 8..15,24..31.
// B fragment (32x16 bf16): lane n&15 holds column N, K = sel*8+0..7, 16+sel*8+0..7 (same as A).
// C/D (16x16 f32, 8 VGPR): vgpr r, lane l -> row = sel*8+r, col = l&15.
// ---------------------------------------------------------------------------
__global__ void gemm_nt_wmma_bf16(const __bf16* __restrict__ A,
                                  const __bf16* __restrict__ B,
                                  const float* __restrict__ bias,
                                  float* __restrict__ C,
                                  int M, int N, int K) {
  const int wave = (blockIdx.x * blockDim.x + threadIdx.x) >> 5;
  const int lane = threadIdx.x & 31;
  const int ntn  = N >> 4;
  const int tm   = wave / ntn;
  const int tn   = wave - tm * ntn;
  if (tm >= (M >> 4)) return;                  // wave-uniform guard

  const int m   = lane & 15;
  const int sel = lane >> 4;

  const __bf16* __restrict__ arow = A + (size_t)(tm * 16 + m) * K;
  const __bf16* __restrict__ brow = B + (size_t)(tn * 16 + m) * K;

  v8f c = {};
  for (int kb = 0; kb < K; kb += 32) {
    // A: two contiguous 16B chunks per lane (interleaved K layout)
    v8bf a0 = *(const v8bf*)(arow + kb + sel * 8);
    v8bf a1 = *(const v8bf*)(arow + kb + 16 + sel * 8);
    // B: two contiguous 16B chunks per lane (interleaved K layout, matches A)
    v8bf b0 = *(const v8bf*)(brow + kb + sel * 8);
    v8bf b1 = *(const v8bf*)(brow + kb + 16 + sel * 8);
    v16bf av, bv;
#pragma unroll
    for (int i = 0; i < 8; ++i) {
      av[i] = a0[i]; av[i + 8] = a1[i];
      bv[i] = b0[i]; bv[i + 8] = b1[i];
    }
    c = __builtin_amdgcn_wmma_f32_16x16x32_bf16(
        /*neg_a=*/false, av, /*neg_b=*/false, bv,
        /*c_mod=*/(short)0, c, /*reuse_a=*/false, /*reuse_b=*/false);
  }

  const int col = tn * 16 + m;
  const float bval = bias ? bias[col] : 0.0f;
#pragma unroll
  for (int r = 0; r < 8; ++r) {
    const int row = tm * 16 + sel * 8 + r;
    C[(size_t)row * N + col] = c[r] + bval;
  }
}

// ---------------------------------------------------------------------------
// Fused ring-buffer write + single-query attention (HBM-streaming phase).
// Layout: each lane owns a float4 of the 256-float K/V row (b128 loads),
// 2 waves cover a full row; a lane's 4 dims belong to one head
// (head = half*4 + lane/8), so the q.k dot is 4 FMAs + a 3-step xor-shuffle
// reduce over an aligned 8-lane group. Online softmax per head-group; the
// accumulator is a float4 per lane. 4 batches per 256-thread block.
// wm_K / wm_V streamed once with non-temporal b128 loads (1 GB, no reuse).
// ---------------------------------------------------------------------------
__global__ void __launch_bounds__(256)
attention_stream(const float* __restrict__ q,
                 const float* __restrict__ knew,
                 const float* __restrict__ vnew,
                 const float* __restrict__ wm_K,
                 const float* __restrict__ wm_V,
                 const unsigned char* __restrict__ reset_mask,
                 const unsigned char* __restrict__ wm_valid,
                 const long long* __restrict__ wm_ptr,
                 __bf16* __restrict__ attn_out) {
  const int wib  = threadIdx.x >> 5;            // wave in block: 0..7
  const int lane = threadIdx.x & 31;
  const int b    = blockIdx.x * 4 + (wib >> 1); // 4 batches per block
  const int half = wib & 1;                     // which 512B half of the row
  const int head = half * 4 + (lane >> 3);      // 0..7 (uniform per 8 lanes)
  const int d0   = half * 128 + lane * 4;       // this lane's 4 dims

  const bool keep = (reset_mask[b] == 0);
  const int  ptr  = keep ? (int)wm_ptr[b] : 0;

  const v4f   q4    = *(const v4f*)(q + (size_t)b * DWM + d0);
  const float slope = exp2f(-(float)(head + 1));   // 2^(-8*(h+1)/H), H=8

  const float* __restrict__ Kbase = wm_K + (size_t)b * WSLOT * DWM;
  const float* __restrict__ Vbase = wm_V + (size_t)b * WSLOT * DWM;
  const unsigned char* __restrict__ valrow = wm_valid + (size_t)b * WSLOT;

  float mrun = -INFINITY, srun = 0.0f;
  v4f acc = {0.0f, 0.0f, 0.0f, 0.0f};

  for (int j = 0; j < WSLOT; ++j) {
    const bool is_ptr = (j == ptr);
    const bool valid  = is_ptr || (keep && valrow[j] != 0);
    if (!valid) continue;                         // wave-uniform branch

    if (j + 8 < WSLOT) {                          // stream-ahead prefetch
      __builtin_prefetch(Kbase + (size_t)(j + 8) * DWM + d0, 0, 0);
      __builtin_prefetch(Vbase + (size_t)(j + 8) * DWM + d0, 0, 0);
    }

    v4f k4, v4;
    if (is_ptr) {                                 // ring-buffer written slot
      k4 = *(const v4f*)(knew + (size_t)b * DWM + d0);
      v4 = *(const v4f*)(vnew + (size_t)b * DWM + d0);
    } else {                                      // non-temporal HBM stream
      k4 = __builtin_nontemporal_load((const v4f*)(Kbase + (size_t)j * DWM + d0));
      v4 = __builtin_nontemporal_load((const v4f*)(Vbase + (size_t)j * DWM + d0));
    }

    // partial dot over this lane's 4 dims, then reduce over the 8-lane head
    float part = q4.x * k4.x;
    part = fmaf(q4.y, k4.y, part);
    part = fmaf(q4.z, k4.z, part);
    part = fmaf(q4.w, k4.w, part);
    part += __shfl_xor(part, 4, 32);
    part += __shfl_xor(part, 2, 32);
    part += __shfl_xor(part, 1, 32);              // all 8 lanes hold the dot

    const int   dist  = (ptr - j + WSLOT) & (WSLOT - 1);   // (ptr-j) mod W
    const float score = part * ATT_SCALE - slope * (float)dist;

    // online softmax update (replicated within the 8-lane head group)
    const float mnew   = fmaxf(mrun, score);
    const float rescal = __expf(mrun - mnew);     // exp(-inf)=0 on first hit
    const float p      = __expf(score - mnew);
    srun  = srun * rescal + p;
    acc.x = fmaf(p, v4.x, acc.x * rescal);
    acc.y = fmaf(p, v4.y, acc.y * rescal);
    acc.z = fmaf(p, v4.z, acc.z * rescal);
    acc.w = fmaf(p, v4.w, acc.w * rescal);
    mrun = mnew;
  }

  // slot `ptr` is always valid -> srun > 0
  const float inv = 1.0f / srun;
  v4bf o;
  o[0] = (__bf16)(acc.x * inv);
  o[1] = (__bf16)(acc.y * inv);
  o[2] = (__bf16)(acc.z * inv);
  o[3] = (__bf16)(acc.w * inv);
  *(v4bf*)(attn_out + (size_t)b * DWM + d0) = o;
}

// ---------------------------------------------------------------------------
// Host-side orchestration (graph-capture safe: only kernel launches on stream)
// ---------------------------------------------------------------------------
extern "C" void kernel_launch(void* const* d_in, const int* in_sizes, int n_in,
                              void* d_out, int out_size, void* d_ws, size_t ws_size,
                              hipStream_t stream) {
  const float*         x       = (const float*)d_in[0];
  const float*         wm_K    = (const float*)d_in[1];
  const float*         wm_V    = (const float*)d_in[2];
  const float*         Wq_w    = (const float*)d_in[3];
  const float*         Wq_b    = (const float*)d_in[4];
  const float*         Wk_w    = (const float*)d_in[5];
  const float*         Wk_b    = (const float*)d_in[6];
  const float*         Wv_w    = (const float*)d_in[7];
  const float*         Wv_b    = (const float*)d_in[8];
  const float*         Wo_w    = (const float*)d_in[9];
  const float*         Wo_b    = (const float*)d_in[10];
  const unsigned char* reset   = (const unsigned char*)d_in[11];
  const unsigned char* valid   = (const unsigned char*)d_in[12];
  const long long*     wm_ptr  = (const long long*)d_in[13];
  float*               y       = (float*)d_out;

  // Workspace carve-out (256B-aligned slabs)
  char* ws = (char*)d_ws;
  size_t off = 0;
  auto carve = [&](size_t bytes) -> void* {
    void* p = ws + off;
    off += (bytes + 255) & ~(size_t)255;
    return p;
  };
  __bf16* x_bf  = (__bf16*)carve((size_t)BSZ * DIM * sizeof(__bf16));   // 2 MB
  __bf16* wq_bf = (__bf16*)carve((size_t)DWM * DIM * sizeof(__bf16));   // 512 KB
  __bf16* wk_bf = (__bf16*)carve((size_t)DWM * DIM * sizeof(__bf16));
  __bf16* wv_bf = (__bf16*)carve((size_t)DWM * DIM * sizeof(__bf16));
  __bf16* wo_bf = (__bf16*)carve((size_t)DIM * DWM * sizeof(__bf16));
  float*  qbuf  = (float*)carve((size_t)BSZ * DWM * sizeof(float));     // 1 MB
  float*  kbuf  = (float*)carve((size_t)BSZ * DWM * sizeof(float));
  float*  vbuf  = (float*)carve((size_t)BSZ * DWM * sizeof(float));
  __bf16* ao_bf = (__bf16*)carve((size_t)BSZ * DWM * sizeof(__bf16));   // 512 KB

  // 1) bf16 staging of GEMM operands
  {
    const int nx = BSZ * DIM, nw = DWM * DIM;
    cvt_f32_bf16<<<(nx + 255) / 256, 256, 0, stream>>>(x,    x_bf,  nx);
    cvt_f32_bf16<<<(nw + 255) / 256, 256, 0, stream>>>(Wq_w, wq_bf, nw);
    cvt_f32_bf16<<<(nw + 255) / 256, 256, 0, stream>>>(Wk_w, wk_bf, nw);
    cvt_f32_bf16<<<(nw + 255) / 256, 256, 0, stream>>>(Wv_w, wv_bf, nw);
    cvt_f32_bf16<<<(nw + 255) / 256, 256, 0, stream>>>(Wo_w, wo_bf, nw);
  }

  // 2) q/k/v projections: C[BSZ,DWM] = x * W^T + b   (WMMA bf16)
  {
    const int tiles  = (BSZ / 16) * (DWM / 16);   // 1024 waves
    const int blocks = tiles / 4;                 // 4 waves / 128-thread block
    gemm_nt_wmma_bf16<<<blocks, 128, 0, stream>>>(x_bf, wq_bf, Wq_b, qbuf, BSZ, DWM, DIM);
    gemm_nt_wmma_bf16<<<blocks, 128, 0, stream>>>(x_bf, wk_bf, Wk_b, kbuf, BSZ, DWM, DIM);
    gemm_nt_wmma_bf16<<<blocks, 128, 0, stream>>>(x_bf, wv_bf, Wv_b, vbuf, BSZ, DWM, DIM);
  }

  // 3) Fused ring-write + attention stream (bandwidth-bound: 1 GB K/V)
  attention_stream<<<BSZ / 4, 256, 0, stream>>>(
      qbuf, kbuf, vbuf, wm_K, wm_V, reset, valid, wm_ptr, ao_bf);

  // 4) Output projection: y[BSZ,DIM] = attn_out * Wo^T + b  (WMMA bf16)
  {
    const int tiles  = (BSZ / 16) * (DIM / 16);   // 4096 waves
    const int blocks = tiles / 4;
    gemm_nt_wmma_bf16<<<blocks, 128, 0, stream>>>(ao_bf, wo_bf, Wo_b, y, BSZ, DIM, DWM);
  }
}
